// Attention_21543555957314
// MI455X (gfx1250) — matrix operand order repaired and checked by reading, not compiled, not run on hardware
//
#include <hip/hip_runtime.h>
#include <math.h>

#define HD 1024
#define SEQ 32768
#define BM 128
#define LDA 1032  // padded LDS row (bf16 elems) to break bank-conflict pattern

typedef __attribute__((ext_vector_type(16))) __bf16 v16bf;
typedef __attribute__((ext_vector_type(8)))  float  v8f;

struct __align__(16) U4 { unsigned x, y, z, w; };
struct __align__(16) Frag32 { U4 lo, hi; };   // 32B = one v16bf operand

// fp32 -> bf16, round-to-nearest-even
__device__ __forceinline__ unsigned short f2bf(float x) {
    unsigned u = __float_as_uint(x);
    u += 0x7FFFu + ((u >> 16) & 1u);
    return (unsigned short)(u >> 16);
}

// c[j] = attn_b[j] + sum_{k<2H} hidden[k] * attn_w[j*3H + k]
__global__ void prep_c(const float* __restrict__ hidden,
                       const float* __restrict__ attn_w,
                       const float* __restrict__ attn_b,
                       float* __restrict__ cvec) {
    int j = blockIdx.x * 256 + threadIdx.x;       // 0..1023
    const float* wr = attn_w + (size_t)j * (3 * HD);
    float s = attn_b[j];
    for (int k = 0; k < 2 * HD; k += 4) {
        float4 h = *(const float4*)(hidden + k);
        float4 w = *(const float4*)(wr + k);
        s += h.x * w.x + h.y * w.y + h.z * w.z + h.w * w.w;
    }
    cvec[j] = s;
}

// W2bf[j*H + k] = bf16(attn_w[j*3H + 2H + k]), row-major H x H
__global__ void prep_w2(const float* __restrict__ attn_w,
                        unsigned short* __restrict__ w2) {
    int idx = (blockIdx.x * 256 + threadIdx.x) * 4;   // < H*H
    int j = idx >> 10, k = idx & (HD - 1);
    float4 f = *(const float4*)(attn_w + (size_t)j * (3 * HD) + 2 * HD + k);
    uint2 pk;
    pk.x = (unsigned)f2bf(f.x) | ((unsigned)f2bf(f.y) << 16);
    pk.y = (unsigned)f2bf(f.z) | ((unsigned)f2bf(f.w) << 16);
    *(uint2*)(w2 + idx) = pk;
}

// Fused: logits[s] = sum_j v[j] * tanh(c[j] + enc[s,:] . W2[j,:])
__global__ void attn_gemm(const float* __restrict__ enc,
                          const unsigned short* __restrict__ w2,
                          const float* __restrict__ cvec,
                          const float* __restrict__ vw,
                          float* __restrict__ logits) {
    extern __shared__ unsigned short As[];        // [BM][LDA] bf16
    const int tid = threadIdx.x;                  // 256 threads = 8 waves
    const int block_row = blockIdx.x * BM;

    // ---- Stage A block: 128 x 1024 fp32 -> bf16 in LDS (coalesced b128 loads)
    for (int i = tid; i < BM * (HD / 4); i += 256) {
        int row = i >> 8;                         // 256 float4 per row
        int c4  = (i & 255) << 2;
        float4 f = *(const float4*)(enc + (size_t)(block_row + row) * HD + c4);
        uint2 pk;
        pk.x = (unsigned)f2bf(f.x) | ((unsigned)f2bf(f.y) << 16);
        pk.y = (unsigned)f2bf(f.z) | ((unsigned)f2bf(f.w) << 16);
        *(uint2*)(As + row * LDA + c4) = pk;
    }
    __syncthreads();

    const int wid  = tid >> 5;                    // m-tile 0..7
    const int lane = tid & 31;
    const int half = lane >> 4;                   // K-half per 16-bit layouts
    const int l    = lane & 15;                   // M row (A) / N col (B)

    // A: lane half 0 holds K {0..7,16..23}, half 1 holds K {8..15,24..31}
    const unsigned short* a_base = As + (wid * 16 + l) * LDA + half * 8;
    // B: lane n holds column j=n_base+l; half 0 holds K {0..7,16..23}, half 1 holds K {8..15,24..31}
    const unsigned short* b_lane = w2 + (size_t)l * HD + half * 8;

    float pacc[8];
#pragma unroll
    for (int r = 0; r < 8; ++r) pacc[r] = 0.f;

    for (int ng = 0; ng < 16; ++ng) {             // 64 N-tiles, 4 per group
        v8f acc[4];
#pragma unroll
        for (int t = 0; t < 4; ++t) acc[t] = v8f{0.f,0.f,0.f,0.f,0.f,0.f,0.f,0.f};
        const unsigned short* bp = b_lane + (size_t)(ng * 64) * HD;

        for (int kk = 0; kk < 32; ++kk) {         // K = 1024 in steps of 32
            const int k0 = kk * 32;
            Frag32 fa;
            fa.lo = *(const U4*)(a_base + k0);        // K chunk 0 (8 bf16)
            fa.hi = *(const U4*)(a_base + k0 + 16);   // K chunk 1 (8 bf16)
            v16bf a = __builtin_bit_cast(v16bf, fa);
#pragma unroll
            for (int t = 0; t < 4; ++t) {
                const unsigned short* p = bp + (size_t)(t * 16) * HD + k0;
                Frag32 fb;
                fb.lo = *(const U4*)(p);
                fb.hi = *(const U4*)(p + 16);
                v16bf b = __builtin_bit_cast(v16bf, fb);
                acc[t] = __builtin_amdgcn_wmma_f32_16x16x32_bf16(
                    false, a, false, b, (short)0, acc[t], false, false);
            }
        }
        // Epilogue: + c[j], tanh, * v[j], accumulate per-row partial sums
#pragma unroll
        for (int t = 0; t < 4; ++t) {
            int j = ng * 64 + t * 16 + l;
            float cj = cvec[j];
            float vj = vw[j];
#pragma unroll
            for (int r = 0; r < 8; ++r) {
                float e = tanhf(acc[t][r] + cj);
                pacc[r] += vj * e;
            }
        }
    }

    // Reduce across the 16 lanes holding each row (C/D layout: half owns rows r / r+8)
#pragma unroll
    for (int r = 0; r < 8; ++r) {
        float v = pacc[r];
        v += __shfl_xor(v, 1);
        v += __shfl_xor(v, 2);
        v += __shfl_xor(v, 4);
        v += __shfl_xor(v, 8);
        pacc[r] = v;
    }
    if (l == 0) {
        int rb = block_row + wid * 16 + half * 8;
#pragma unroll
        for (int r = 0; r < 8; ++r) logits[rb + r] = pacc[r];
    }
}

__global__ void softmax_k(const float* __restrict__ lg, float* __restrict__ out) {
    __shared__ float sm[32];
    __shared__ float bcast;
    const int tid = threadIdx.x;                  // 1024 threads = 32 waves
    float m = -3.4e38f;
    for (int i = tid; i < SEQ; i += 1024) m = fmaxf(m, lg[i]);
    for (int o = 16; o > 0; o >>= 1) m = fmaxf(m, __shfl_xor(m, o));
    if ((tid & 31) == 0) sm[tid >> 5] = m;
    __syncthreads();
    if (tid < 32) {
        float v = sm[tid];
        for (int o = 16; o > 0; o >>= 1) v = fmaxf(v, __shfl_xor(v, o));
        if (tid == 0) bcast = v;
    }
    __syncthreads();
    float gm = bcast;
    float s = 0.f;
    for (int i = tid; i < SEQ; i += 1024) s += __expf(lg[i] - gm);
    for (int o = 16; o > 0; o >>= 1) s += __shfl_xor(s, o);
    __syncthreads();
    if ((tid & 31) == 0) sm[tid >> 5] = s;
    __syncthreads();
    if (tid < 32) {
        float v = sm[tid];
        for (int o = 16; o > 0; o >>= 1) v += __shfl_xor(v, o);
        if (tid == 0) bcast = v;
    }
    __syncthreads();
    float inv = 1.f / bcast;
    for (int i = tid; i < SEQ; i += 1024) out[i] = __expf(lg[i] - gm) * inv;
}

extern "C" void kernel_launch(void* const* d_in, const int* in_sizes, int n_in,
                              void* d_out, int out_size, void* d_ws, size_t ws_size,
                              hipStream_t stream) {
    const float* hidden = (const float*)d_in[0];   // [1, 2H]
    const float* enc    = (const float*)d_in[1];   // [S, H]
    const float* attn_w = (const float*)d_in[2];   // [H, 3H]
    const float* attn_b = (const float*)d_in[3];   // [H]
    const float* v_w    = (const float*)d_in[4];   // [1, H]
    float* out = (float*)d_out;                    // [S]

    char* ws = (char*)d_ws;
    unsigned short* w2bf = (unsigned short*)ws;                 // H*H bf16 = 2 MB
    float* cvec   = (float*)(ws + (size_t)HD * HD * 2);         // H fp32
    float* logits = cvec + HD;                                  // S fp32

    prep_c <<<HD / 256, 256, 0, stream>>>(hidden, attn_w, attn_b, cvec);
    prep_w2<<<(HD * HD / 4) / 256, 256, 0, stream>>>(attn_w, w2bf);

    size_t smem = (size_t)BM * LDA * sizeof(unsigned short);    // 258 KB
    attn_gemm<<<SEQ / BM, 256, smem, stream>>>(enc, w2bf, cvec, v_w, logits);

    softmax_k<<<1, 1024, 0, stream>>>(logits, out);
}
